// ModelParallelSoftmaxLoss_10746008174736
// MI455X (gfx1250) — matrix operand order repaired and checked by reading, not compiled, not run on hardware
//
#include <hip/hip_runtime.h>
#include <hip/hip_bf16.h>
#include <stdint.h>

#define NROWS 4096
#define DDIM  512
#define VDIM  100000

#define ROWS_PER_BLOCK 128      // 8 waves * 16 rows
#define CTILE 64                // vocab columns per LDS stage (4 x 16)
#define CHUNK 4096              // vocab columns per block
#define NCHUNK 25               // ceil(100000/4096)
#define ITERS (CHUNK / CTILE)   // 64
#define KSTEPS (DDIM / 32)      // 16 wmma k-steps

#define LDS_ROW_BYTES 1040      // 512*2 + 16 pad (16B aligned, breaks bank clash)
#define LDS_ROW_U32   260
#define LDS_BUF_BYTES (CTILE * LDS_ROW_BYTES)   // 66,560 B per stage buffer

typedef __attribute__((ext_vector_type(16))) __bf16 v16bf;
typedef __attribute__((ext_vector_type(8)))  float  v8f;
typedef int v4i_vec __attribute__((vector_size(16)));
typedef __attribute__((address_space(1))) v4i_vec* gbl_v4i_ptr;
typedef __attribute__((address_space(3))) v4i_vec* lds_v4i_ptr;

union FragBF { uint32_t u[8]; v16bf v; };

__device__ __forceinline__ uint32_t pack_bf16x2(float a, float b) {
  uint32_t ua = __float_as_uint(a);
  uint32_t ub = __float_as_uint(b);
  ua += 0x7FFFu + ((ua >> 16) & 1u);   // round-to-nearest-even
  ub += 0x7FFFu + ((ub >> 16) & 1u);
  return (ua >> 16) | (ub & 0xFFFF0000u);
}

__device__ __forceinline__ float rmax16(float v) {
  v = fmaxf(v, __shfl_xor(v, 1, 32));
  v = fmaxf(v, __shfl_xor(v, 2, 32));
  v = fmaxf(v, __shfl_xor(v, 4, 32));
  v = fmaxf(v, __shfl_xor(v, 8, 32));
  return v;
}
__device__ __forceinline__ float rsum16(float v) {
  v += __shfl_xor(v, 1, 32);
  v += __shfl_xor(v, 2, 32);
  v += __shfl_xor(v, 4, 32);
  v += __shfl_xor(v, 8, 32);
  return v;
}

__device__ __forceinline__ void async_cp16(void* l, const void* g) {
#if __has_builtin(__builtin_amdgcn_global_load_async_to_lds_b128)
  __builtin_amdgcn_global_load_async_to_lds_b128(
      (gbl_v4i_ptr)g, (lds_v4i_ptr)l, 0, 0);
#else
  uint32_t lo = (uint32_t)(uintptr_t)l;
  uint64_t ga = (uint64_t)(uintptr_t)g;
  asm volatile("global_load_async_to_lds_b128 %0, %1, off"
               :: "v"(lo), "v"(ga) : "memory");
#endif
}
template <int N>
__device__ __forceinline__ void wait_asyncN() {
#if __has_builtin(__builtin_amdgcn_s_wait_asynccnt)
  __builtin_amdgcn_s_wait_asynccnt(N);
#else
  asm volatile("s_wait_asynccnt %0" :: "i"(N) : "memory");
#endif
}

// ---------------- conversion: f32 pairs -> packed bf16x2 ----------------
__global__ void cvt_f32_to_bf16x2(const float2* __restrict__ src,
                                  uint32_t* __restrict__ dst, int count) {
  int i = blockIdx.x * blockDim.x + threadIdx.x;
  int stride = gridDim.x * blockDim.x;
  for (; i < count; i += stride) {
    float2 v = src[i];
    dst[i] = pack_bf16x2(v.x, v.y);
  }
}

// ---------------- fused logits GEMM + online logsumexp ----------------
template <bool PRECONV>
__global__ __launch_bounds__(256)
void fused_logits_lse(const uint32_t* __restrict__ xb,   // x bf16x2 [N][D/2]
                      const uint32_t* __restrict__ wb,   // W bf16x2 [V][D/2] (PRECONV)
                      const float*    __restrict__ W,    // W f32 (fallback)
                      const float*    __restrict__ bias, // [V]
                      float* __restrict__ pm,            // [N][NCHUNK]
                      float* __restrict__ ps) {          // [N][NCHUNK]
  // double-buffered W staging: 2 x 65KB (well under 320KB/WGP LDS)
  __shared__ __align__(16) uint8_t ldsW[2 * LDS_BUF_BYTES];

  const int tid   = threadIdx.x;
  const int lane  = tid & 31;
  const int wave  = tid >> 5;
  const int half  = lane >> 4;
  const int qlane = lane & 15;

  const int rowBase   = blockIdx.x * ROWS_PER_BLOCK + wave * 16;
  const int chunkBase = blockIdx.y * CHUNK;

  // preload A fragments: this wave's 16 rows, all 512 K (128 VGPRs, reused 64x)
  FragBF aF[KSTEPS];
  {
    const int row = rowBase + qlane;                    // A: M = lane%16
    const uint32_t* xr = xb + (size_t)row * (DDIM / 2);
    #pragma unroll
    for (int ks = 0; ks < KSTEPS; ++ks) {
      #pragma unroll
      for (int j = 0; j < 8; ++j) {
        // A 16x32, contiguous k order used here (B matches): Vj -> K=2j (+16*half)
        int kh = ks * 16 + j + (half << 3); // = k/2
        aF[ks].u[j] = xr[kh];
      }
    }
  }

  float mrun[8], srun[8];
  #pragma unroll
  for (int r = 0; r < 8; ++r) { mrun[r] = -1e30f; srun[r] = 0.0f; }

  // issue 16 async b128 copies moving one 64-row W tile into LDS buffer `buf`
  auto fillAsync = [&](int buf, int vb) {
    uint8_t* base = ldsW + buf * LDS_BUF_BYTES;
    #pragma unroll
    for (int i = 0; i < 16; ++i) {
      int e   = tid + i * 256;                          // 0..4095
      int rr  = e >> 6;
      int pos = e & 63;
      int v   = vb + rr; v = (v < VDIM) ? v : (VDIM - 1);
      async_cp16(base + rr * LDS_ROW_BYTES + pos * 16,
                 (const uint8_t*)wb + (size_t)v * (DDIM * 2) + pos * 16);
    }
  };

  // synchronous fallback fill (f32 load + convert + ds_store) into buffer 0
  auto fillSync = [&](int vb) {
    uint32_t* ldsU = (uint32_t*)ldsW;
    #pragma unroll
    for (int i = 0; i < 64; ++i) {
      int e   = tid + i * 256;                          // 0..16383
      int rr  = e >> 8;
      int pos = e & 255;                                // float2 index within row
      int v   = vb + rr; v = (v < VDIM) ? v : (VDIM - 1);
      float2 w2 = ((const float2*)(W + (size_t)v * DDIM))[pos];
      ldsU[rr * LDS_ROW_U32 + pos] = pack_bf16x2(w2.x, w2.y);
    }
  };

  // 64 WMMAs on buffer `buf`, then fold the 16x64 tile into running (m, s)
  auto computeFold = [&](int buf, int vb) {
    const uint32_t* ldsU = (const uint32_t*)(ldsW + buf * LDS_BUF_BYTES);
    v8f acc[4] = {};
    #pragma unroll
    for (int ks = 0; ks < KSTEPS; ++ks) {
      #pragma unroll
      for (int t = 0; t < 4; ++t) {
        FragBF bF;
        // B 32x16: lane%16 = N; contiguous k order used here (A matches): Vj -> K=2j (+16*half)
        const uint32_t* bl = ldsU + (t * 16 + qlane) * LDS_ROW_U32 + ks * 16 + half * 8;
        #pragma unroll
        for (int j = 0; j < 8; ++j) bF.u[j] = bl[j];
        acc[t] = __builtin_amdgcn_wmma_f32_16x16x32_bf16(
            false, aF[ks].v, false, bF.v, (short)0, acc[t], false, false);
      }
    }
    // C layout: lane%16 = N, vgpr r -> M = r + 8*half
    float bvs[4]; bool bok[4];
    #pragma unroll
    for (int t = 0; t < 4; ++t) {
      int col = vb + t * 16 + qlane;
      int cc  = (col < VDIM) ? col : (VDIM - 1);
      bvs[t] = bias[cc];
      bok[t] = (col < VDIM);
    }
    #pragma unroll
    for (int r = 0; r < 8; ++r) {
      float v0 = bok[0] ? (acc[0][r] + bvs[0]) : -1e30f;
      float v1 = bok[1] ? (acc[1][r] + bvs[1]) : -1e30f;
      float v2 = bok[2] ? (acc[2][r] + bvs[2]) : -1e30f;
      float v3 = bok[3] ? (acc[3][r] + bvs[3]) : -1e30f;
      float mx = rmax16(fmaxf(fmaxf(v0, v1), fmaxf(v2, v3)));
      float mnew = fmaxf(mrun[r], mx);
      float se = __expf(v0 - mnew) + __expf(v1 - mnew) +
                 __expf(v2 - mnew) + __expf(v3 - mnew);
      se = rsum16(se);
      srun[r] = srun[r] * __expf(mrun[r] - mnew) + se;
      mrun[r] = mnew;
    }
  };

  if (PRECONV) {
    // software pipeline: DMA of tile it+1 overlaps WMMA of tile it
    fillAsync(0, chunkBase);
    for (int it = 0; it < ITERS; ++it) {
      const int cur = it & 1;
      if (it + 1 < ITERS) {
        fillAsync(1 - cur, chunkBase + (it + 1) * CTILE);
        wait_asyncN<16>();   // in-order ASYNCcnt: <=16 outstanding => stage `it` landed
      } else {
        wait_asyncN<0>();
      }
      __syncthreads();       // stage `it` visible to all waves
      computeFold(cur, chunkBase + it * CTILE);
      __syncthreads();       // all reads of buf[cur] done before it is refilled
    }
  } else {
    for (int it = 0; it < ITERS; ++it) {
      __syncthreads();
      fillSync(chunkBase + it * CTILE);
      __syncthreads();
      computeFold(0, chunkBase + it * CTILE);
    }
  }

  // one writer per (half, r): all 16 lanes of a half hold identical reduced values
  #pragma unroll
  for (int r = 0; r < 8; ++r) {
    if (qlane == r) {
      int row = rowBase + half * 8 + r;
      pm[(size_t)row * NCHUNK + blockIdx.y] = mrun[r];
      ps[(size_t)row * NCHUNK + blockIdx.y] = srun[r];
    }
  }
}

// ---------------- merge chunk partials -> logz[n] ----------------
__global__ void lse_merge(const float* __restrict__ pm, const float* __restrict__ ps,
                          float* __restrict__ logz) {
  int row  = blockIdx.x * 8 + (threadIdx.x >> 5);
  int lane = threadIdx.x & 31;
  if (row >= NROWS) return;
  float m = -1e30f, s = 0.0f;
  if (lane < NCHUNK) {
    m = pm[(size_t)row * NCHUNK + lane];
    s = ps[(size_t)row * NCHUNK + lane];
  }
  float M = m;
  M = fmaxf(M, __shfl_xor(M, 16, 32));
  M = fmaxf(M, __shfl_xor(M, 8, 32));
  M = fmaxf(M, __shfl_xor(M, 4, 32));
  M = fmaxf(M, __shfl_xor(M, 2, 32));
  M = fmaxf(M, __shfl_xor(M, 1, 32));
  float t = s * __expf(m - M);
  t += __shfl_xor(t, 16, 32);
  t += __shfl_xor(t, 8, 32);
  t += __shfl_xor(t, 4, 32);
  t += __shfl_xor(t, 2, 32);
  t += __shfl_xor(t, 1, 32);
  if (lane == 0) logz[row] = M + logf(t);
}

// ---------------- exact f32 target logit: dot(x[n], W[lb[n]]) + b ----------------
__global__ void target_logit(const float* __restrict__ x, const long long* __restrict__ lb,
                             const float* __restrict__ W, const float* __restrict__ bias,
                             float* __restrict__ tgt) {
  int row  = blockIdx.x * 8 + (threadIdx.x >> 5);
  int lane = threadIdx.x & 31;
  if (row >= NROWS) return;
  long long v = lb[row];
  const float* xr = x + (size_t)row * DDIM;
  const float* wr = W + (size_t)v * DDIM;
  float s = 0.0f;
  #pragma unroll
  for (int i = 0; i < DDIM / 32; ++i) s = fmaf(xr[lane + 32 * i], wr[lane + 32 * i], s);
  s += __shfl_xor(s, 16, 32);
  s += __shfl_xor(s, 8, 32);
  s += __shfl_xor(s, 4, 32);
  s += __shfl_xor(s, 2, 32);
  s += __shfl_xor(s, 1, 32);
  if (lane == 0) tgt[row] = s + bias[v];
}

// ---------------- mean(logz - tgt) ----------------
__global__ void loss_reduce(const float* __restrict__ logz, const float* __restrict__ tgt,
                            float* __restrict__ out) {
  __shared__ float red[256];
  float a = 0.0f;
  for (int n = threadIdx.x; n < NROWS; n += 256) a += logz[n] - tgt[n];
  red[threadIdx.x] = a;
  __syncthreads();
  for (int o = 128; o > 0; o >>= 1) {
    if ((int)threadIdx.x < o) red[threadIdx.x] += red[threadIdx.x + o];
    __syncthreads();
  }
  if (threadIdx.x == 0) out[0] = red[0] / (float)NROWS;
}

extern "C" void kernel_launch(void* const* d_in, const int* in_sizes, int n_in,
                              void* d_out, int out_size, void* d_ws, size_t ws_size,
                              hipStream_t stream) {
  const float*     x    = (const float*)d_in[0];
  const long long* lb   = (const long long*)d_in[1];   // int64 labels
  const float*     W    = (const float*)d_in[2];
  const float*     bias = (const float*)d_in[3];
  float* out = (float*)d_out;
  char*  ws  = (char*)d_ws;

  size_t off = 0;
  uint32_t* xb   = (uint32_t*)(ws + off); off += (size_t)NROWS * DDIM * 2;    // 4 MB
  float*    pm   = (float*)(ws + off);    off += (size_t)NROWS * NCHUNK * 4;  // 400 KB
  float*    ps   = (float*)(ws + off);    off += (size_t)NROWS * NCHUNK * 4;  // 400 KB
  float*    logz = (float*)(ws + off);    off += (size_t)NROWS * 4;
  float*    tgt  = (float*)(ws + off);    off += (size_t)NROWS * 4;
  size_t wb_bytes = (size_t)VDIM * DDIM * 2;                                  // 102.4 MB
  uint32_t* wbq = (uint32_t*)(ws + off);
  bool preconv = (ws_size >= off + wb_bytes);   // bf16 W fits in L2 (192MB); f32 doesn't

  {
    int cnt = NROWS * DDIM / 2;
    cvt_f32_to_bf16x2<<<cnt / 256, 256, 0, stream>>>((const float2*)x, xb, cnt);
  }
  if (preconv) {
    int cnt = (int)((size_t)VDIM * DDIM / 2);
    cvt_f32_to_bf16x2<<<8192, 256, 0, stream>>>((const float2*)W, wbq, cnt);
  }

  dim3 grid(NROWS / ROWS_PER_BLOCK, NCHUNK);
  if (preconv)
    fused_logits_lse<true ><<<grid, 256, 0, stream>>>(xb, wbq, W, bias, pm, ps);
  else
    fused_logits_lse<false><<<grid, 256, 0, stream>>>(xb, wbq, W, bias, pm, ps);

  lse_merge   <<<NROWS / 8, 256, 0, stream>>>(pm, ps, logz);
  target_logit<<<NROWS / 8, 256, 0, stream>>>(x, lb, W, bias, tgt);
  loss_reduce <<<1, 256, 0, stream>>>(logz, tgt, out);
}
